// SelfAttentionModel_39848706572877
// MI455X (gfx1250) — matrix operand order repaired and checked by reading, not compiled, not run on hardware
//
#include <hip/hip_runtime.h>
#include <hip/hip_bf16.h>

// ---- problem constants (match reference) ----
#define B_ 2
#define S_ 2048
#define H_ 16
#define E_ 64
// scores are divided by 8 -> fold 1/8 into Wq/bq (projection), so qh = q/8.

typedef _Float16 v16h __attribute__((ext_vector_type(16)));
typedef _Float16 v8h  __attribute__((ext_vector_type(8)));
typedef float    v8f  __attribute__((ext_vector_type(8)));

static __device__ __forceinline__ v16h pack16(v8h lo, v8h hi8) {
    v16h r;
#pragma unroll
    for (int j = 0; j < 8; ++j) { r[j] = lo[j]; r[j + 8] = hi8[j]; }
    return r;
}

static __device__ __forceinline__ v8f wmma_f16(v16h a, v16h b, v8f c) {
    // D = A(16x32 f16) * B(32x16 f16) + C(16x16 f32)
    return __builtin_amdgcn_wmma_f32_16x16x32_f16(
        /*neg_a=*/false, a, /*neg_b=*/false, b,
        /*c_mod=*/(short)0, c, /*reuse_a=*/false, /*reuse_b=*/false);
}

// Convert two contiguous 8-float groups into a f16 operand register.
static __device__ __forceinline__ v16h cvt16(const float* g0, const float* g1) {
    const v8f lo = *(const v8f*)g0;
    const v8f hv = *(const v8f*)g1;
    v16h r;
#pragma unroll
    for (int j = 0; j < 8; ++j) {
        r[j]     = (_Float16)lo[j];
        r[j + 8] = (_Float16)hv[j];
    }
    return r;
}

// =====================================================================
// Kernel 0: one-shot weight conversion to f16 (Wq pre-scaled by 1/8).
//   3 x 64 x 64 elements; runs once, removes per-block W conversion.
// =====================================================================
__global__ __launch_bounds__(256) void convert_w_kernel(
    const float* __restrict__ Wq, const float* __restrict__ Wk,
    const float* __restrict__ Wv, _Float16* __restrict__ wh)
{
    const int i = blockIdx.x * 256 + threadIdx.x;   // 0 .. 3*4096-1
    const int m = i >> 12;                          // matrix id
    const int e = i & 4095;
    float v;
    if (m == 0)      v = Wq[e] * 0.125f;
    else if (m == 1) v = Wk[e];
    else             v = Wv[e];
    wh[i] = (_Float16)v;
}

// =====================================================================
// Kernel 1: WMMA QKV projection. One wave32 projects 16 flat rows
// (H=16 -> a 16-row tile is exactly (b,s,h=0..15)) for q, k and v.
//   A: input rows, f32->f16 in-register
//   B: pre-converted f16 W rows (B-layout K contiguous = W's e index)
//   C: bias broadcast (lane owns one output column)
// Outputs: qh[b,h,s,e] (=q/8), kh[b,h,s,e], vt[b,h,e,s]  (all f16)
// =====================================================================
__global__ __launch_bounds__(32) void proj_qkv_kernel(
    const float* __restrict__ query, const float* __restrict__ key,
    const float* __restrict__ value,
    const _Float16* __restrict__ wh,
    const float* __restrict__ bq, const float* __restrict__ bk,
    const float* __restrict__ bv,
    _Float16* __restrict__ qh, _Float16* __restrict__ kh,
    _Float16* __restrict__ vt)
{
    const int lane = threadIdx.x & 31;
    const int hi   = lane >> 4;
    const int lp   = lane & 15;

    const int r0 = blockIdx.x * 16;        // flat row base; h = row-in-tile
    const int b  = r0 / (S_ * H_);
    const int s  = (r0 / H_) % S_;

    const _Float16* wqh = wh;              // [64][64] f16, Wq/8
    const _Float16* wkh = wh + 4096;
    const _Float16* wvh = wh + 8192;

    // ---- input A operands (16 rows x 64 K, f32 -> f16) ----
    // A layout: element j holds K = k0 + 16*hi + j
    const float* qrow = query + (size_t)(r0 + lp) * E_;
    const float* krow = key   + (size_t)(r0 + lp) * E_;
    const float* vrow = value + (size_t)(r0 + lp) * E_;

    const v16h aQ0 = cvt16(qrow + 16 * hi,      qrow + 16 * hi + 8);
    const v16h aQ1 = cvt16(qrow + 32 + 16 * hi, qrow + 40 + 16 * hi);
    const v16h aK0 = cvt16(krow + 16 * hi,      krow + 16 * hi + 8);
    const v16h aK1 = cvt16(krow + 32 + 16 * hi, krow + 40 + 16 * hi);
    const v16h aV0 = cvt16(vrow + 16 * hi,      vrow + 16 * hi + 8);
    const v16h aV1 = cvt16(vrow + 32 + 16 * hi, vrow + 40 + 16 * hi);

#pragma unroll
    for (int ft = 0; ft < 4; ++ft) {            // 16-wide feature tiles
        const int f = lp + 16 * ft;             // this lane's output column
        const _Float16* wq = wqh + (size_t)f * E_ + 16 * hi;
        const _Float16* wk = wkh + (size_t)f * E_ + 16 * hi;
        const _Float16* wv = wvh + (size_t)f * E_ + 16 * hi;
        const v16h bWq0 = pack16(*(const v8h*)(wq),      *(const v8h*)(wq + 8));
        const v16h bWq1 = pack16(*(const v8h*)(wq + 32), *(const v8h*)(wq + 40));
        const v16h bWk0 = pack16(*(const v8h*)(wk),      *(const v8h*)(wk + 8));
        const v16h bWk1 = pack16(*(const v8h*)(wk + 32), *(const v8h*)(wk + 40));
        const v16h bWv0 = pack16(*(const v8h*)(wv),      *(const v8h*)(wv + 8));
        const v16h bWv1 = pack16(*(const v8h*)(wv + 32), *(const v8h*)(wv + 40));

        const float cq = bq[f] * 0.125f;
        const float ck = bk[f];
        const float cv = bv[f];
        v8f accq = (v8f){cq, cq, cq, cq, cq, cq, cq, cq};
        v8f acck = (v8f){ck, ck, ck, ck, ck, ck, ck, ck};
        v8f accv = (v8f){cv, cv, cv, cv, cv, cv, cv, cv};

        accq = wmma_f16(aQ0, bWq0, accq);
        accq = wmma_f16(aQ1, bWq1, accq);
        acck = wmma_f16(aK0, bWk0, acck);
        acck = wmma_f16(aK1, bWk1, acck);
        accv = wmma_f16(aV0, bWv0, accv);
        accv = wmma_f16(aV1, bWv1, accv);

        // D layout: lane owns column f, VGPR g holds row h = g + 8*hi
#pragma unroll
        for (int g = 0; g < 8; ++g) {
            const int h  = g + 8 * hi;
            const int bh = b * H_ + h;
            qh[((size_t)bh * S_ + s) * E_ + f] = (_Float16)accq[g];
            kh[((size_t)bh * S_ + s) * E_ + f] = (_Float16)acck[g];
            vt[((size_t)bh * E_ + f) * S_ + s] = (_Float16)accv[g];
        }
    }
}

// =====================================================================
// Kernel 2: flash attention, one wave32 per 32-query tile (2 A tiles).
//   grid = (S/32, B*H), block = 32
// =====================================================================
__global__ __launch_bounds__(32) void flash_attn_kernel(
    const _Float16* __restrict__ qh, const _Float16* __restrict__ kh,
    const _Float16* __restrict__ vt, const int* __restrict__ attn_mask,
    float* __restrict__ out)
{
    __shared__ __align__(32) float Slds[32][32];   // masked scores (f32)
    __shared__ __align__(32) float Alds[32];       // per-row alpha / row-sum

    const int lane = threadIdx.x & 31;
    const int hi   = lane >> 4;        // half-wave select
    const int lp   = lane & 15;        // row (A) / column (B,C,D) within tile
    const int bh   = blockIdx.y;       // 0 .. B*H-1
    const int b    = bh >> 4;
    const int h    = bh & 15;
    const int s0   = blockIdx.x * 32;  // query tile base (32 rows)

    // ---- load Q tiles as A operands: aQ[mt][k0/32] ----
    v16h aQ[2][2];
#pragma unroll
    for (int mt = 0; mt < 2; ++mt) {
        const _Float16* qrow =
            qh + ((size_t)bh * S_ + (s0 + mt * 16 + lp)) * E_;
        aQ[mt][0] = pack16(*(const v8h*)(qrow +  0 + 16 * hi),
                           *(const v8h*)(qrow +  8 + 16 * hi));
        aQ[mt][1] = pack16(*(const v8h*)(qrow + 32 + 16 * hi),
                           *(const v8h*)(qrow + 40 + 16 * hi));
    }

    // ---- running state ----
    v8f   O[2][4];                      // O[mt][et]: rows mt*16+g+8*hi, cols lp+16*et
    float rm[2] = {-1e30f, -1e30f};
    float rl[2] = {0.0f, 0.0f};

#pragma unroll
    for (int mt = 0; mt < 2; ++mt)
#pragma unroll
        for (int et = 0; et < 4; ++et)
            O[mt][et] = (v8f){0, 0, 0, 0, 0, 0, 0, 0};

    const int* mrow = attn_mask + (size_t)b * S_;

    for (int kc = 0; kc < S_; kc += 32) {
        // ============ phase 1: scores (D layout) =========================
        const _Float16* kb0 = kh + ((size_t)bh * S_ + (kc + lp)) * E_;
        const _Float16* kb1 = kh + ((size_t)bh * S_ + (kc + 16 + lp)) * E_;
        if (kc + 32 < S_)
            __builtin_prefetch(kb0 + 32 * E_, 0, 0);   // global_prefetch_b8

        v16h bK[2][2];                                  // [nt][k0/32]
        bK[0][0] = pack16(*(const v8h*)(kb0 + 16 * hi),
                          *(const v8h*)(kb0 + 16 * hi + 8));
        bK[0][1] = pack16(*(const v8h*)(kb0 + 32 + 16 * hi),
                          *(const v8h*)(kb0 + 32 + 16 * hi + 8));
        bK[1][0] = pack16(*(const v8h*)(kb1 + 16 * hi),
                          *(const v8h*)(kb1 + 16 * hi + 8));
        bK[1][1] = pack16(*(const v8h*)(kb1 + 32 + 16 * hi),
                          *(const v8h*)(kb1 + 32 + 16 * hi + 8));

        const int m0 = mrow[kc + lp];
        const int m1 = mrow[kc + 16 + lp];

        v8f sc[2][2];                                   // [mt][nt]
#pragma unroll
        for (int mt = 0; mt < 2; ++mt) {
#pragma unroll
            for (int nt = 0; nt < 2; ++nt) {
                v8f s = (v8f){0, 0, 0, 0, 0, 0, 0, 0};
                s = wmma_f16(aQ[mt][0], bK[nt][0], s);
                s = wmma_f16(aQ[mt][1], bK[nt][1], s);
                sc[mt][nt] = s;
            }
        }

        // masked scores -> LDS (lane = column n; rows mt*16 + g + 8*hi)
#pragma unroll
        for (int mt = 0; mt < 2; ++mt)
#pragma unroll
            for (int g = 0; g < 8; ++g) {
                const int r = mt * 16 + g + 8 * hi;
                Slds[r][lp]      = m0 ? sc[mt][0][g] : -1e30f;
                Slds[r][16 + lp] = m1 ? sc[mt][1][g] : -1e30f;
            }
        __syncthreads();

        // ============ phase 2: softmax in row ownership ==================
        v16h aP[2];
#pragma unroll
        for (int mt = 0; mt < 2; ++mt) {
            const int r = mt * 16 + lp;
            const v8f lo = *(const v8f*)(&Slds[r][16 * hi]);
            const v8f hv = *(const v8f*)(&Slds[r][16 * hi + 8]);

            float mx = lo[0];
#pragma unroll
            for (int j = 1; j < 8; ++j) mx = fmaxf(mx, lo[j]);
#pragma unroll
            for (int j = 0; j < 8; ++j) mx = fmaxf(mx, hv[j]);
            mx = fmaxf(mx, __shfl_xor(mx, 16, 32));    // combine row halves

            const float nm    = fmaxf(rm[mt], mx);
            const float alpha = __expf(rm[mt] - nm);
            rm[mt] = nm;

            float ls = 0.0f;
#pragma unroll
            for (int j = 0; j < 8; ++j) {
                const float p = (lo[j] > -1e29f) ? __expf(lo[j] - nm) : 0.0f;
                aP[mt][j] = (_Float16)p;
                ls += p;
            }
#pragma unroll
            for (int j = 0; j < 8; ++j) {
                const float p = (hv[j] > -1e29f) ? __expf(hv[j] - nm) : 0.0f;
                aP[mt][8 + j] = (_Float16)p;
                ls += p;
            }
            ls += __shfl_xor(ls, 16, 32);              // full row sum
            rl[mt] = rl[mt] * alpha + ls;

            Alds[r] = alpha;   // lanes lp and lp+16 write identical value
        }
        __syncthreads();

        // ============ phase 3: rescale O and accumulate P*V ==============
        v8f av[2];
#pragma unroll
        for (int mt = 0; mt < 2; ++mt)
            av[mt] = *(const v8f*)(&Alds[mt * 16 + 8 * hi]);

#pragma unroll
        for (int mt = 0; mt < 2; ++mt)
#pragma unroll
            for (int et = 0; et < 4; ++et)
#pragma unroll
                for (int g = 0; g < 8; ++g) O[mt][et][g] *= av[mt][g];

        v16h bV[4];
#pragma unroll
        for (int et = 0; et < 4; ++et) {
            const _Float16* vrowp =
                vt + ((size_t)bh * E_ + (lp + 16 * et)) * S_ + kc + 16 * hi;
            bV[et] = pack16(*(const v8h*)vrowp, *(const v8h*)(vrowp + 8));
        }
#pragma unroll
        for (int et = 0; et < 4; ++et)
#pragma unroll
            for (int mt = 0; mt < 2; ++mt)
                O[mt][et] = wmma_f16(aP[mt], bV[et], O[mt][et]);
    }

    // ---- finalize: divide by row sums, store fp32 [B,S,H,E] ----
    __syncthreads();
#pragma unroll
    for (int mt = 0; mt < 2; ++mt) Alds[mt * 16 + lp] = rl[mt];
    __syncthreads();

#pragma unroll
    for (int mt = 0; mt < 2; ++mt) {
        const v8f lv = *(const v8f*)(&Alds[mt * 16 + 8 * hi]);
        float inv[8];
#pragma unroll
        for (int g = 0; g < 8; ++g)
            inv[g] = (lv[g] > 0.0f) ? (1.0f / lv[g]) : 0.0f;
#pragma unroll
        for (int et = 0; et < 4; ++et)
#pragma unroll
            for (int g = 0; g < 8; ++g) {
                const int m = mt * 16 + g + 8 * hi;
                const int e = lp + 16 * et;
                out[(((size_t)b * S_ + (s0 + m)) * H_ + h) * E_ + e] =
                    O[mt][et][g] * inv[g];
            }
    }
}

// =====================================================================
// Host launcher
// =====================================================================
extern "C" void kernel_launch(void* const* d_in, const int* in_sizes, int n_in,
                              void* d_out, int out_size, void* d_ws, size_t ws_size,
                              hipStream_t stream) {
    const float* query = (const float*)d_in[0];
    const float* key_  = (const float*)d_in[1];
    const float* value = (const float*)d_in[2];
    const int*   amask = (const int*)  d_in[3];
    const float* Wq    = (const float*)d_in[4];
    const float* bq    = (const float*)d_in[5];
    const float* Wk    = (const float*)d_in[6];
    const float* bk    = (const float*)d_in[7];
    const float* Wv    = (const float*)d_in[8];
    const float* bv    = (const float*)d_in[9];
    float* out = (float*)d_out;

    const size_t tsz = (size_t)B_ * H_ * S_ * E_;   // elements per f16 tensor
    _Float16* qh = (_Float16*)d_ws;                 // 8 MB
    _Float16* kh = qh + tsz;                        // 8 MB
    _Float16* vt = kh + tsz;                        // 8 MB
    _Float16* wh = vt + tsz;                        // 24 KB (3 x 64 x 64 f16)

    convert_w_kernel<<<dim3((3 * E_ * E_) / 256), 256, 0, stream>>>(
        Wq, Wk, Wv, wh);

    proj_qkv_kernel<<<dim3((B_ * S_ * H_) / 16), 32, 0, stream>>>(
        query, key_, value, wh, bq, bk, bv, qh, kh, vt);

    flash_attn_kernel<<<dim3(S_ / 32, B_ * H_), 32, 0, stream>>>(
        qh, kh, vt, amask, out);
}
